// GroupedQueryAttentionWithRoPEAndCache_3753801417336
// MI455X (gfx1250) — matrix operand order repaired and checked by reading, not compiled, not run on hardware
//
#include <hip/hip_runtime.h>
#include <hip/hip_bf16.h>

// ---------------------------------------------------------------------------
// Grouped-Query Attention + RoPE prefill for MI455X (gfx1250, wave32, WMMA).
// All GEMM-like math runs on V_WMMA_F32_16X16X32_F16 (f16 in, f32 accumulate).
// GEMM A-tiles are staged to LDS with GLOBAL_LOAD_ASYNC_TO_LDS_B128 and
// double-buffered (ASYNCcnt pipeline).  Softmax row reductions use
// v_permlane16_b32 (VALU) instead of ds_bpermute.
// ---------------------------------------------------------------------------

#define B_   2
#define S_   2048
#define E_   4096
#define HQ_  32
#define HKV_ 8
#define D_   128
#define SCALE_F (1.0f / 64.0f)   // 1/sqrt(E)

#define USE_ASYNC_LDS 1

typedef __attribute__((ext_vector_type(16))) _Float16 v16h;
typedef __attribute__((ext_vector_type(8)))  float    v8f;
typedef __attribute__((ext_vector_type(4)))  unsigned int u32x4;

// Load a 16-half WMMA fragment as two contiguous 16-byte chunks.
__device__ __forceinline__ v16h ld_frag(const _Float16* p0, const _Float16* p1) {
  union { u32x4 u[2]; v16h h; } t;
  t.u[0] = *(const u32x4*)p0;
  t.u[1] = *(const u32x4*)p1;
  return t.h;
}

// 16-byte global -> LDS copy (async on CDNA5; tracked by ASYNCcnt).
__device__ __forceinline__ void cp_g2l_16B(_Float16* lptr, const _Float16* gptr) {
#if USE_ASYNC_LDS
  asm volatile("global_load_async_to_lds_b128 %0, %1, off"
               :: "v"((unsigned)(uintptr_t)lptr),
                  "v"((unsigned long long)(uintptr_t)gptr)
               : "memory");
#else
  *(u32x4*)lptr = *(const u32x4*)gptr;
#endif
}
__device__ __forceinline__ void cp_g2l_wait() {
#if USE_ASYNC_LDS
  asm volatile("s_wait_asynccnt 0" ::: "memory");
#endif
}

// ---- 16-lane-row butterfly reductions (VALU permlane, no LDS traffic) ----
#if __has_builtin(__builtin_amdgcn_permlane16)
__device__ __forceinline__ float plane16(float v, unsigned s0, unsigned s1) {
  unsigned u = __builtin_bit_cast(unsigned, v);
  u = __builtin_amdgcn_permlane16(u, u, s0, s1, false, false);
  return __builtin_bit_cast(float, u);
}
__device__ __forceinline__ float row16_max(float v) {
  v = fmaxf(v, plane16(v, 0x67452301u, 0xEFCDAB89u));  // xor 1
  v = fmaxf(v, plane16(v, 0x54761032u, 0xDCFE98BAu));  // xor 2
  v = fmaxf(v, plane16(v, 0x32107654u, 0xBA98FEDCu));  // xor 4
  v = fmaxf(v, plane16(v, 0xFEDCBA98u, 0x76543210u));  // xor 8
  return v;
}
__device__ __forceinline__ float row16_sum(float v) {
  v += plane16(v, 0x67452301u, 0xEFCDAB89u);
  v += plane16(v, 0x54761032u, 0xDCFE98BAu);
  v += plane16(v, 0x32107654u, 0xBA98FEDCu);
  v += plane16(v, 0xFEDCBA98u, 0x76543210u);
  return v;
}
#else
__device__ __forceinline__ float row16_max(float v) {
  for (int off = 8; off >= 1; off >>= 1) v = fmaxf(v, __shfl_xor(v, off, 32));
  return v;
}
__device__ __forceinline__ float row16_sum(float v) {
  for (int off = 8; off >= 1; off >>= 1) v += __shfl_xor(v, off, 32);
  return v;
}
#endif

// ---------------------------------------------------------------------------
// 1a) Weight transpose + f32->f16 convert: src (K x N) f32 -> dst (N x K) f16
// ---------------------------------------------------------------------------
__global__ void transpose_cvt_kernel(const float* __restrict__ src,
                                     _Float16* __restrict__ dst,
                                     int K, int N) {
  __shared__ float tile[32][33];
  const int kBase = blockIdx.y * 32;
  const int nBase = blockIdx.x * 32;
  const int tx = threadIdx.x;          // 0..31
  const int ty = threadIdx.y;          // 0..7
#pragma unroll
  for (int i = 0; i < 32; i += 8)
    tile[ty + i][tx] = src[(size_t)(kBase + ty + i) * N + nBase + tx];
  __syncthreads();
#pragma unroll
  for (int i = 0; i < 32; i += 8)
    dst[(size_t)(nBase + ty + i) * K + kBase + tx] = (_Float16)tile[tx][ty + i];
}

// ---------------------------------------------------------------------------
// 1b) Plain f32 -> f16 convert (for x), 8 elements / thread
// ---------------------------------------------------------------------------
__global__ void cvt_f16_kernel(const float* __restrict__ src,
                               _Float16* __restrict__ dst, size_t n) {
  const size_t i = ((size_t)blockIdx.x * blockDim.x + threadIdx.x) * 8;
  if (i >= n) return;
  const float4* p = (const float4*)(src + i);
  float4 a = p[0], b = p[1];
  union { u32x4 u; _Float16 h[8]; } t;
  t.h[0] = (_Float16)a.x; t.h[1] = (_Float16)a.y;
  t.h[2] = (_Float16)a.z; t.h[3] = (_Float16)a.w;
  t.h[4] = (_Float16)b.x; t.h[5] = (_Float16)b.y;
  t.h[6] = (_Float16)b.z; t.h[7] = (_Float16)b.w;
  *(u32x4*)(dst + i) = t.u;
}

// ---------------------------------------------------------------------------
// 2) Tiled WMMA GEMM:  C(MxN) = A(MxK, f16 row-major) * BT(NxK, f16)^T
//    Block tile 128x128, 8 waves of 64x32.  A double-buffered in LDS via
//    async copies; B fragments straight from global (K-contiguous per column,
//    hot in the 192MB L2) with prefetch of the next K-step.
// ---------------------------------------------------------------------------
template <bool OUT_HALF>
__global__ void __launch_bounds__(256)
gemm_tn_kernel(const _Float16* __restrict__ A, const _Float16* __restrict__ BT,
               void* __restrict__ C, int M, int N, int K, int ldc) {
  __shared__ _Float16 Alds[2][128][48];          // ping-pong 32-K tiles
  const int mBase = blockIdx.y * 128;
  const int nBase = blockIdx.x * 128;
  const int t    = threadIdx.x;
  const int lane = t & 31;
  const int w    = t >> 5;
  const int wm   = (w >> 2) * 64;                // wave M offset (0 / 64)
  const int wn   = (w & 3) * 32;                 // wave N offset (0..96)
  const int l16  = lane & 15;
  const bool hi  = lane >= 16;
  const int kOffA = hi ? 8 : 0;                  // A-frag K start within lane
  const int kloB  = hi ? 16 : 0;                 // unused; B frags also start at kOffA

  v8f acc[4][2] = {};

  const int rowA = t >> 1;                       // 0..127
  const int khA  = (t & 1) * 16;                 // 0 / 16
  const _Float16* aRow = A + (size_t)(mBase + rowA) * K + khA;

  // preload K-tile 0
  cp_g2l_16B(&Alds[0][rowA][khA],     aRow);
  cp_g2l_16B(&Alds[0][rowA][khA + 8], aRow + 8);
  cp_g2l_wait();
  __syncthreads();

  int buf = 0;
  for (int k0 = 0; k0 < K; k0 += 32) {
    // ---- issue async copy of the next tile while computing this one ----
    if (k0 + 32 < K) {
      const _Float16* nxt = aRow + k0 + 32;
      cp_g2l_16B(&Alds[buf ^ 1][rowA][khA],     nxt);
      cp_g2l_16B(&Alds[buf ^ 1][rowA][khA + 8], nxt + 8);
    }

    // ---- fragments + WMMA on current buffer ----
    v16h af[4];
#pragma unroll
    for (int mt = 0; mt < 4; ++mt) {
      const _Float16* base = &Alds[buf][wm + mt * 16 + l16][0];
      af[mt] = ld_frag(base + kOffA, base + 16 + kOffA);
    }
#pragma unroll
    for (int nt = 0; nt < 2; ++nt) {
      const _Float16* bp =
          BT + (size_t)(nBase + wn + nt * 16 + l16) * K + k0 + kOffA;
      v16h bf = ld_frag(bp, bp + 16);
      __builtin_prefetch(bp + 32, 0, 3);         // next K-step (global_prefetch)
#pragma unroll
      for (int mt = 0; mt < 4; ++mt)
        acc[mt][nt] = __builtin_amdgcn_wmma_f32_16x16x32_f16(
            false, af[mt], false, bf, (short)0, acc[mt][nt], false, false);
    }

    cp_g2l_wait();        // own async copies landed
    __syncthreads();      // everyone's copies + reads of current buf done
    buf ^= 1;
  }

  // ---- epilogue: C layout -> lane holds col l16, rows r / r+8 ----
#pragma unroll
  for (int mt = 0; mt < 4; ++mt)
#pragma unroll
    for (int nt = 0; nt < 2; ++nt)
#pragma unroll
      for (int r = 0; r < 8; ++r) {
        const int gr = mBase + wm + mt * 16 + r + (hi ? 8 : 0);
        const int gc = nBase + wn + nt * 16 + l16;
        if (OUT_HALF)
          ((_Float16*)C)[(size_t)gr * ldc + gc] = (_Float16)acc[mt][nt][r];
        else
          ((float*)C)[(size_t)gr * ldc + gc] = acc[mt][nt][r];
      }
}

// ---------------------------------------------------------------------------
// 3) RoPE + layout:  qkv_raw (B*S x 6144 f16)  ->
//      q_attn (B,HQ,S,D) f16 (scaled by 1/64),
//      k_attn (B,HKV,S,D) f16,
//      v_attn (B,HKV,D,S) f16  (d-major so PV B-fragments are key-contiguous)
// ---------------------------------------------------------------------------
__global__ void rope_layout_kernel(const _Float16* __restrict__ qkv,
                                   const float* __restrict__ fc,
                                   _Float16* __restrict__ q_attn,
                                   _Float16* __restrict__ k_attn,
                                   _Float16* __restrict__ v_attn) {
  const size_t idx = (size_t)blockIdx.x * blockDim.x + threadIdx.x;
  if (idx >= (size_t)(B_ * S_) * 3072) return;
  const int cp = (int)(idx % 3072);
  const int m  = (int)(idx / 3072);          // 0..4095 = b*S + s
  const int b  = m >> 11;
  const int s  = m & (S_ - 1);
  const int j  = cp * 2;
  const float e0 = (float)qkv[(size_t)m * 6144 + j];
  const float e1 = (float)qkv[(size_t)m * 6144 + j + 1];
  if (j < 4096) {                             // Q + RoPE + fold softmax scale
    const int h = j >> 7, d = j & 127, p = d >> 1;
    const float c  = fc[((size_t)s * 64 + p) * 2];
    const float sn = fc[((size_t)s * 64 + p) * 2 + 1];
    const size_t o = (((size_t)(b * HQ_ + h) * S_) + s) * D_ + d;
    q_attn[o]     = (_Float16)((e0 * c - e1 * sn) * SCALE_F);
    q_attn[o + 1] = (_Float16)((e0 * sn + e1 * c) * SCALE_F);
  } else if (j < 5120) {                      // K + RoPE
    const int jj = j - 4096;
    const int h = jj >> 7, d = jj & 127, p = d >> 1;
    const float c  = fc[((size_t)s * 64 + p) * 2];
    const float sn = fc[((size_t)s * 64 + p) * 2 + 1];
    const size_t o = (((size_t)(b * HKV_ + h) * S_) + s) * D_ + d;
    k_attn[o]     = (_Float16)(e0 * c - e1 * sn);
    k_attn[o + 1] = (_Float16)(e0 * sn + e1 * c);
  } else {                                    // V: transpose to (d, s)
    const int jj = j - 5120;
    const int h = jj >> 7, d = jj & 127;
    const size_t o = (((size_t)(b * HKV_ + h) * D_) + d) * S_ + s;
    v_attn[o]      = (_Float16)e0;
    v_attn[o + S_] = (_Float16)e1;            // d+1 row
  }
}

// ---------------------------------------------------------------------------
// 4) Flash attention (causal, GQA 4:1).  One wave = one 16-row Q tile.
//    Interior chunks unmasked; one diagonal chunk applies the causal mask.
// ---------------------------------------------------------------------------
template <bool MASKED>
__device__ __forceinline__ void attn_chunk(
    int kc, int qBase, bool hi, int l16,
    const _Float16* __restrict__ kbase, const _Float16* __restrict__ vbase,
    const v16h (&qf)[4], v8f (&out)[8], float (&mrow)[8], float (&lrow)[8],
    _Float16 (*Pw)[48], int kOffA, int kloB) {
  v8f sc[2] = {};
#pragma unroll
  for (int st = 0; st < 2; ++st) {
    const int key = kc + st * 16 + l16;       // this lane's key column
    const _Float16* kp = kbase + (size_t)key * D_;
#pragma unroll
    for (int f = 0; f < 4; ++f) {
      v16h kf = ld_frag(kp + f * 32 + kOffA, kp + f * 32 + 16 + kOffA);
      sc[st] = __builtin_amdgcn_wmma_f32_16x16x32_f16(
          false, qf[f], false, kf, (short)0, sc[st], false, false);
    }
    if (!MASKED)
      __builtin_prefetch(kp + (size_t)32 * D_, 0, 3);   // next chunk's K rows
    if (MASKED) {
#pragma unroll
      for (int r = 0; r < 8; ++r) {           // causal mask (diagonal only)
        const int rq = qBase + r + (hi ? 8 : 0);
        if (key > rq) sc[st][r] = -3.0e38f;
      }
    }
  }
  // online softmax per row (row = 16 lanes of a half-wave)
#pragma unroll
  for (int r = 0; r < 8; ++r) {
    const float mx = row16_max(fmaxf(sc[0][r], sc[1][r]));
    const float mnew = fmaxf(mrow[r], mx);
    const float resc = __expf(mrow[r] - mnew);
    const float p0 = __expf(sc[0][r] - mnew);
    const float p1 = __expf(sc[1][r] - mnew);
    lrow[r] = lrow[r] * resc + row16_sum(p0 + p1);
    mrow[r] = mnew;
#pragma unroll
    for (int dt = 0; dt < 8; ++dt) out[dt][r] *= resc;
    sc[0][r] = p0; sc[1][r] = p1;
  }
  // P (C layout) -> LDS -> A fragment layout (in-wave, DS in-order)
#pragma unroll
  for (int r = 0; r < 8; ++r) {
    const int rl = r + (hi ? 8 : 0);
    Pw[rl][l16]      = (_Float16)sc[0][r];
    Pw[rl][16 + l16] = (_Float16)sc[1][r];
  }
  asm volatile("s_wait_dscnt 0" ::: "memory");
  const _Float16* pb = &Pw[l16][0];
  v16h pf = ld_frag(pb + kOffA, pb + 16 + kOffA);
#pragma unroll
  for (int dt = 0; dt < 8; ++dt) {
    const _Float16* vp = vbase + (size_t)(dt * 16 + l16) * S_ + kc + kOffA;
    v16h vf = ld_frag(vp, vp + 16);
    if (!MASKED) __builtin_prefetch(vp + 32, 0, 3);     // next chunk's V keys
    out[dt] = __builtin_amdgcn_wmma_f32_16x16x32_f16(
        false, pf, false, vf, (short)0, out[dt], false, false);
  }
}

__global__ void __launch_bounds__(256)
flash_attn_kernel(const _Float16* __restrict__ q_attn,
                  const _Float16* __restrict__ k_attn,
                  const _Float16* __restrict__ v_attn,
                  _Float16* __restrict__ attn_out) {
  __shared__ _Float16 Plds[8][16][48];        // per-wave P tile (16 x 32, padded)
  const int lane = threadIdx.x & 31;
  const int w    = threadIdx.x >> 5;
  const int qt   = blockIdx.x * 8 + w;        // 0..127
  const int qBase = qt * 16;
  const int h  = blockIdx.y;
  const int b  = blockIdx.z;
  const int hk = h >> 2;                      // KV_REP = 4
  const int l16 = lane & 15;
  const bool hi = lane >= 16;
  const int kOffA = hi ? 8 : 0;
  const int kloB  = hi ? 16 : 0;

  // Q fragments (resident): lane = query row, d contiguous
  const _Float16* qptr =
      q_attn + (((size_t)(b * HQ_ + h) * S_) + qBase + l16) * D_;
  v16h qf[4];
#pragma unroll
  for (int f = 0; f < 4; ++f)
    qf[f] = ld_frag(qptr + f * 32 + kOffA, qptr + f * 32 + 16 + kOffA);

  v8f out[8] = {};
  float mrow[8], lrow[8];
#pragma unroll
  for (int r = 0; r < 8; ++r) { mrow[r] = -3.0e38f; lrow[r] = 0.0f; }

  const _Float16* kbase = k_attn + (size_t)(b * HKV_ + hk) * S_ * D_;
  const _Float16* vbase = v_attn + (size_t)(b * HKV_ + hk) * D_ * S_;
  _Float16 (*Pw)[48] = Plds[w];

  // unmasked interior chunks: keys kc..kc+31 all <= qBase (min row)
  const int nfull = qBase >> 5;
  for (int c = 0; c < nfull; ++c)
    attn_chunk<false>(c * 32, qBase, hi, l16, kbase, vbase,
                      qf, out, mrow, lrow, Pw, kOffA, kloB);
  // single diagonal chunk with causal mask
  attn_chunk<true>(nfull * 32, qBase, hi, l16, kbase, vbase,
                   qf, out, mrow, lrow, Pw, kOffA, kloB);

  // normalize + store f16 to (B*S, HQ*D) for the output projection
#pragma unroll
  for (int dt = 0; dt < 8; ++dt)
#pragma unroll
    for (int r = 0; r < 8; ++r) {
      const int rq = qBase + r + (hi ? 8 : 0);
      attn_out[((size_t)(b * S_ + rq)) * (HQ_ * D_) + h * D_ + dt * 16 + l16] =
          (_Float16)(out[dt][r] / lrow[r]);
    }
}

// ---------------------------------------------------------------------------
// Host launcher
// ---------------------------------------------------------------------------
extern "C" void kernel_launch(void* const* d_in, const int* in_sizes, int n_in,
                              void* d_out, int out_size, void* d_ws, size_t ws_size,
                              hipStream_t stream) {
  (void)in_sizes; (void)n_in; (void)out_size; (void)ws_size;
  const float* x  = (const float*)d_in[0];
  const float* fc = (const float*)d_in[2];   // freqs_cis (S, D/2, 2)
  const float* wq = (const float*)d_in[3];
  const float* wk = (const float*)d_in[4];
  const float* wv = (const float*)d_in[5];
  const float* wo = (const float*)d_in[6];

  char* ws = (char*)d_ws;
  size_t off = 0;
  auto take = [&](size_t bytes) -> char* {
    char* p = ws + off;
    off = (off + bytes + 255) & ~(size_t)255;
    return p;
  };
  _Float16* wqkvT   = (_Float16*)take((size_t)6144 * 4096 * 2);   // fused B^T
  _Float16* woT     = (_Float16*)take((size_t)4096 * 4096 * 2);
  _Float16* x16     = (_Float16*)take((size_t)4096 * 4096 * 2);
  _Float16* qkv_raw = (_Float16*)take((size_t)4096 * 6144 * 2);
  _Float16* q_attn  = (_Float16*)take((size_t)B_ * HQ_  * S_ * D_ * 2);
  _Float16* k_attn  = (_Float16*)take((size_t)B_ * HKV_ * S_ * D_ * 2);
  _Float16* v_attn  = (_Float16*)take((size_t)B_ * HKV_ * S_ * D_ * 2);
  _Float16* attn_o  = (_Float16*)take((size_t)4096 * 4096 * 2);

  // 1) weight transpose + cvt (wq|wk|wv fused into one N=6144 B^T); x -> f16
  dim3 tb(32, 8);
  transpose_cvt_kernel<<<dim3(128, 128), tb, 0, stream>>>(wq, wqkvT, 4096, 4096);
  transpose_cvt_kernel<<<dim3(32, 128), tb, 0, stream>>>(
      wk, wqkvT + (size_t)4096 * 4096, 4096, 1024);
  transpose_cvt_kernel<<<dim3(32, 128), tb, 0, stream>>>(
      wv, wqkvT + (size_t)5120 * 4096, 4096, 1024);
  transpose_cvt_kernel<<<dim3(128, 128), tb, 0, stream>>>(wo, woT, 4096, 4096);
  cvt_f16_kernel<<<8192, 256, 0, stream>>>(x, x16, (size_t)4096 * 4096);

  // 2) fused QKV projection: (4096 x 4096) * (4096 x 6144)
  gemm_tn_kernel<true><<<dim3(48, 32), 256, 0, stream>>>(
      x16, wqkvT, qkv_raw, 4096, 6144, 4096, 6144);

  // 3) RoPE + attention layouts
  {
    const size_t total = (size_t)(B_ * S_) * 3072;
    rope_layout_kernel<<<(unsigned)((total + 255) / 256), 256, 0, stream>>>(
        qkv_raw, fc, q_attn, k_attn, v_attn);
  }

  // 4) causal GQA flash attention
  flash_attn_kernel<<<dim3(16, HQ_, B_), 256, 0, stream>>>(
      q_attn, k_attn, v_attn, attn_o);

  // 5) output projection: (4096 x 4096) * (4096 x 4096) -> d_out f32
  gemm_tn_kernel<false><<<dim3(32, 32), 256, 0, stream>>>(
      attn_o, woT, (float*)d_out, 4096, 4096, 4096, 4096);
}
